// TATLayer2_22591527977165
// MI455X (gfx1250) — matrix operand order repaired and checked by reading, not compiled, not run on hardware
//
#include <hip/hip_runtime.h>
#include <cstdint>

typedef __bf16 bf16_t;
typedef __attribute__((ext_vector_type(16))) __bf16 v16bf;
typedef __attribute__((ext_vector_type(8)))  float   v8f;
typedef __attribute__((ext_vector_type(8)))  __bf16 bf16x8;
typedef __attribute__((ext_vector_type(4)))  int     i32x4;

#define N_HEADS   8
#define HIDDEN    256
#define DHEAD     32
#define LOGITMAX  4.605170185988092f

#if defined(__HIP_DEVICE_COMPILE__) && __has_builtin(__builtin_amdgcn_global_load_async_to_lds_b128)
#define HAS_ASYNC_LDS 1
typedef __attribute__((address_space(1))) i32x4 as1_i32x4;
typedef __attribute__((address_space(3))) i32x4 as3_i32x4;
#else
#define HAS_ASYNC_LDS 0
#endif

__device__ inline void wait_async0() {
#if HAS_ASYNC_LDS
#if __has_builtin(__builtin_amdgcn_s_wait_asynccnt)
  __builtin_amdgcn_s_wait_asynccnt(0);
#else
  asm volatile("s_wait_asynccnt 0" ::: "memory");
#endif
#endif
}

// ---------------------------------------------------------------------------
// weight conversion + transpose: src f32 [K,N] -> dst bf16 [N,K]
// ---------------------------------------------------------------------------
__global__ __launch_bounds__(256) void cvt_transpose_kernel(
    const float* __restrict__ src, bf16_t* __restrict__ dst, int K, int N) {
  __shared__ bf16_t tile[32][33];
  const int k0 = blockIdx.x * 32, n0 = blockIdx.y * 32;
  const int tx = threadIdx.x & 31, ty = threadIdx.x >> 5;
#pragma unroll
  for (int i = ty; i < 32; i += 8)
    tile[i][tx] = (bf16_t)src[(long)(k0 + i) * N + n0 + tx];
  __syncthreads();
#pragma unroll
  for (int i = ty; i < 32; i += 8)
    dst[(long)(n0 + i) * K + k0 + tx] = tile[tx][i];
}

// ---------------------------------------------------------------------------
// Tiled GEMM: C[M,N] = act(A[M,K] @ W[K,N] + bias)
//  A: f32, row-remapped.  Wt: bf16 W^T [N,K] row-major.  C: f32 dense.
//  BM=BN=128, BK=32, 256 thr = 8 waves, wave tile 64x32 (8 WMMA / k-step).
// ---------------------------------------------------------------------------
template <bool SILU>
__global__ __launch_bounds__(256) void gemm_kernel(
    const float* __restrict__ A, const bf16_t* __restrict__ Wt,
    const float* __restrict__ bias, float* __restrict__ C,
    int K, int N, int rows_per_b, int stride_b, int base_off) {
  __shared__ bf16_t sA[128][40];   // row-major M x K-slab (80B rows, 16B aligned)
  __shared__ bf16_t sB[128][40];   // N-major:  sB[n][k]

  const int tid  = threadIdx.x;
  const int bm   = blockIdx.x, bn = blockIdx.y;
  const int wid  = tid >> 5,  lane = tid & 31;
  const int l16  = lane & 15, half = lane >> 4;
  const int wm   = wid & 1,   wn   = wid >> 1;

  const v8f vzero = {0.f, 0.f, 0.f, 0.f, 0.f, 0.f, 0.f, 0.f};
  v8f acc[4][2];
#pragma unroll
  for (int i = 0; i < 4; i++)
#pragma unroll
    for (int j = 0; j < 2; j++) acc[i][j] = vzero;

  // A staging: 2 threads per row, 16 f32 each (convert f32->bf16)
  const int arow = tid >> 1;
  const int acol = (tid & 1) * 16;
  const int gr   = bm * 128 + arow;
  const long asrc = (long)base_off + (long)(gr / rows_per_b) * stride_b + (gr % rows_per_b);
  const float* aptr = A + asrc * (long)K + acol;
  // B staging: 2 threads per n-row of W^T, 16 contiguous bf16 each
  const int bnr = tid >> 1;
  const int bkc = (tid & 1) * 16;
  const bf16_t* wptr = Wt + ((long)bn * 128 + bnr) * K + bkc;

  for (int k0 = 0; k0 < K; k0 += 32) {
    // ---- stage B (async LDS copy when available) ----
#if HAS_ASYNC_LDS
    __builtin_amdgcn_global_load_async_to_lds_b128(
        (as1_i32x4*)wptr, (as3_i32x4*)&sB[bnr][bkc], 0, 0);
    __builtin_amdgcn_global_load_async_to_lds_b128(
        (as1_i32x4*)(wptr + 8), (as3_i32x4*)&sB[bnr][bkc + 8], 0, 0);
#else
    {
      bf16x8 w0 = ((const bf16x8*)wptr)[0];
      bf16x8 w1 = ((const bf16x8*)wptr)[1];
      *(bf16x8*)&sB[bnr][bkc]     = w0;
      *(bf16x8*)&sB[bnr][bkc + 8] = w1;
    }
#endif
    // ---- stage A (convert f32 -> bf16) ----
    const float4* ap4 = (const float4*)aptr;
    float4 a0 = ap4[0], a1 = ap4[1], a2 = ap4[2], a3 = ap4[3];
    if (k0 + 32 < K) __builtin_prefetch(aptr + 32, 0, 1);
    bf16_t* sw = &sA[arow][acol];
    sw[0]  = (bf16_t)a0.x; sw[1]  = (bf16_t)a0.y; sw[2]  = (bf16_t)a0.z; sw[3]  = (bf16_t)a0.w;
    sw[4]  = (bf16_t)a1.x; sw[5]  = (bf16_t)a1.y; sw[6]  = (bf16_t)a1.z; sw[7]  = (bf16_t)a1.w;
    sw[8]  = (bf16_t)a2.x; sw[9]  = (bf16_t)a2.y; sw[10] = (bf16_t)a2.z; sw[11] = (bf16_t)a2.w;
    sw[12] = (bf16_t)a3.x; sw[13] = (bf16_t)a3.y; sw[14] = (bf16_t)a3.z; sw[15] = (bf16_t)a3.w;
    wait_async0();
    __syncthreads();

    // ---- fragments ----
    v16bf afrag[4];
#pragma unroll
    for (int mt = 0; mt < 4; mt++) {
      const bf16_t* src = &sA[wm * 64 + mt * 16 + l16][0];
#pragma unroll
      for (int r = 0; r < 8; r++) {
        int kk = half * 16 + r * 2;
        afrag[mt][2 * r]     = src[kk];
        afrag[mt][2 * r + 1] = src[kk + 1];
      }
    }
    v16bf bfrag[2];
#pragma unroll
    for (int nt = 0; nt < 2; nt++) {
      const bf16_t* src = &sB[wn * 32 + nt * 16 + l16][half * 16];
#pragma unroll
      for (int e = 0; e < 16; e++) bfrag[nt][e] = src[e];
    }
#pragma unroll
    for (int mt = 0; mt < 4; mt++)
#pragma unroll
      for (int nt = 0; nt < 2; nt++)
        acc[mt][nt] = __builtin_amdgcn_wmma_f32_16x16x32_bf16(
            false, afrag[mt], false, bfrag[nt], (short)0, acc[mt][nt], false, false);
    __syncthreads();
    aptr += 32;
    wptr += 32;
  }

  // epilogue: bias (+SiLU), store f32
  const long crow0 = (long)bm * 128 + wm * 64;
  const int  ncol0 = bn * 128 + wn * 32;
#pragma unroll
  for (int mt = 0; mt < 4; mt++) {
#pragma unroll
    for (int nt = 0; nt < 2; nt++) {
      int n = ncol0 + nt * 16 + l16;
      float bv = bias[n];
#pragma unroll
      for (int r = 0; r < 8; r++) {
        long m  = crow0 + mt * 16 + r + half * 8;
        float v = acc[mt][nt][r] + bv;
        if constexpr (SILU) v = v / (1.0f + __expf(-v));
        C[m * (long)N + n] = v;
      }
    }
  }
}

// ---------------------------------------------------------------------------
// window-partition + bf16 pack of q (from fin), k (pin[t]), v (pin[t]-pin[2])
// ---------------------------------------------------------------------------
typedef __attribute__((ext_vector_type(4))) __bf16 bf16x4;
__global__ void qkv_prep_kernel(const float* __restrict__ fin,
                                const float* __restrict__ pin,
                                bf16_t* __restrict__ qw, bf16_t* __restrict__ kw,
                                bf16_t* __restrict__ vw, int nw, int H, int W) {
  int idx = blockIdx.x * blockDim.x + threadIdx.x;
  int total = 2 * nw * 128 * (HIDDEN / 4);
  if (idx >= total) return;
  int c4  = idx & 63;
  int l   = (idx >> 6) & 127;
  int wdx = (idx >> 13) % nw;
  int b   = idx / (nw << 13);
  int HW  = H * W;
  int t   = l >> 6, rem = l & 63;
  int ii  = rem >> 3, jj = rem & 7;
  int Wc  = W >> 3;
  int wh  = wdx / Wc, ww = wdx % Wc;
  int pos = (wh * 8 + ii) * W + (ww * 8 + jj);
  long finrow = (long)b * (2 * HW) + (long)t * HW + pos;
  long pinrow = (long)b * (3 * HW) + (long)t * HW + pos;
  long plast  = (long)b * (3 * HW) + (long)2 * HW + pos;
  float4 fq = ((const float4*)(fin + finrow * HIDDEN))[c4];
  float4 fk = ((const float4*)(pin + pinrow * HIDDEN))[c4];
  float4 fl = ((const float4*)(pin + plast * HIDDEN))[c4];
  long orow = ((long)(b * nw + wdx) * 128 + l) * HIDDEN;
  bf16x4 q, k, v;
  q.x = (bf16_t)fq.x; q.y = (bf16_t)fq.y; q.z = (bf16_t)fq.z; q.w = (bf16_t)fq.w;
  k.x = (bf16_t)fk.x; k.y = (bf16_t)fk.y; k.z = (bf16_t)fk.z; k.w = (bf16_t)fk.w;
  v.x = (bf16_t)(fk.x - fl.x); v.y = (bf16_t)(fk.y - fl.y);
  v.z = (bf16_t)(fk.z - fl.z); v.w = (bf16_t)(fk.w - fl.w);
  ((bf16x4*)(qw + orow))[c4] = q;
  ((bf16x4*)(kw + orow))[c4] = k;
  ((bf16x4*)(vw + orow))[c4] = v;
}

// intra-half (16-lane) reductions; wave32 -> masks stay inside a half
__device__ inline float hmax16(float v) {
#pragma unroll
  for (int m = 1; m < 16; m <<= 1) v = fmaxf(v, __shfl_xor(v, m, 32));
  return v;
}
__device__ inline float hsum16(float v) {
#pragma unroll
  for (int m = 1; m < 16; m <<= 1) v += __shfl_xor(v, m, 32);
  return v;
}

// ---------------------------------------------------------------------------
// Attention: one block per (b, window, head); 4 waves, 2 row-blocks each.
// position bias staged through LDS (union with P buffer; pos consumed before
// P is produced).  Window-reverse permutation folded into output store.
// ---------------------------------------------------------------------------
__global__ __launch_bounds__(128) void attn_kernel(
    const bf16_t* __restrict__ qw, const bf16_t* __restrict__ kw,
    const bf16_t* __restrict__ vw, const float* __restrict__ pos_bias,
    const float* __restrict__ logit_scale, float* __restrict__ o_out,
    int nw, int H, int W) {
  __shared__ bf16_t sK[128][40];       // sK[L][d]   (B-frag for scores)
  __shared__ bf16_t sV[32][136];       // sV[d][L]   (B-frag for P@V)
  __shared__ bf16_t sQ[4][16][40];     // per-wave q row-block
  union PosP {
    float  pos[16][132];               // 16x128 position tile (f32)
    bf16_t P[16][136];                 // 16x128 probabilities (bf16)
  };
  __shared__ PosP swp[4];              // per-wave

  const int tid  = threadIdx.x;
  const int wid  = tid >> 5, lane = tid & 31;
  const int l16  = lane & 15, half = lane >> 4;
  const int bid  = blockIdx.x;
  const int h    = bid & 7;
  const int wdx  = (bid >> 3) % nw;
  const int b    = bid / (nw * 8);
  const int co   = h * DHEAD;
  const long rowbase = (long)(b * nw + wdx) * 128;

  // stage K and V^T for this head (coalesced 16B loads)
  {
    const bf16x8* k8 = (const bf16x8*)(kw + (rowbase + tid) * HIDDEN + co);
    const bf16x8* v8 = (const bf16x8*)(vw + (rowbase + tid) * HIDDEN + co);
#pragma unroll
    for (int g = 0; g < 4; g++) {
      bf16x8 kv = k8[g];
      bf16x8 vv = v8[g];
#pragma unroll
      for (int j = 0; j < 8; j++) {
        sK[tid][g * 8 + j] = kv[j];
        sV[g * 8 + j][tid] = vv[j];
      }
    }
  }
  __syncthreads();

  const float sc = __expf(fminf(logit_scale[h], LOGITMAX));
  const int Wc = W >> 3;
  const int wh = wdx / Wc, ww = wdx % Wc;
  const int HW = H * W;
  const v8f vzero = {0.f, 0.f, 0.f, 0.f, 0.f, 0.f, 0.f, 0.f};

  for (int rbi = 0; rbi < 2; rbi++) {
    const int rb  = wid + rbi * 4;
    const int qr0 = rb * 16;
    const float* pbase = pos_bias + (((long)b * N_HEADS + h) * 128 + qr0) * 128;

    // stage 16x128 position tile (coalesced float4 loads)
#pragma unroll
    for (int i = 0; i < 16; i++) {
      float4 t = ((const float4*)(pbase + i * 128))[lane];
      *(float4*)&swp[wid].pos[i][lane * 4] = t;
    }

    // stage this wave's 16x32 q block
    {
      int row = lane >> 1;
      int cb  = (lane & 1) * 16;
      const bf16_t* qr = qw + (rowbase + qr0 + row) * HIDDEN + co + cb;
      bf16x8 q0 = ((const bf16x8*)qr)[0];
      bf16x8 q1 = ((const bf16x8*)qr)[1];
      *(bf16x8*)&sQ[wid][row][cb]     = q0;
      *(bf16x8*)&sQ[wid][row][cb + 8] = q1;
    }

    // q A-fragment
    v16bf aq;
    {
      const bf16_t* src = &sQ[wid][l16][0];
#pragma unroll
      for (int r = 0; r < 8; r++) {
        int kk = half * 16 + r * 2;
        aq[2 * r]     = src[kk];
        aq[2 * r + 1] = src[kk + 1];
      }
    }

    // scores: 16 x 128 (8 WMMA @ K=32)
    v8f s[8];
#pragma unroll
    for (int nt = 0; nt < 8; nt++) {
      v16bf bk_;
      const bf16_t* src = &sK[nt * 16 + l16][half * 16];
#pragma unroll
      for (int e = 0; e < 16; e++) bk_[e] = src[e];
      s[nt] = __builtin_amdgcn_wmma_f32_16x16x32_bf16(false, aq, false, bk_,
                                                      (short)0, vzero, false, false);
    }

    // scale + position bias (from LDS tile)
#pragma unroll
    for (int nt = 0; nt < 8; nt++)
#pragma unroll
      for (int r = 0; r < 8; r++)
        s[nt][r] = s[nt][r] * sc + swp[wid].pos[r + 8 * half][nt * 16 + l16];

    // softmax per row (row = r + 8*half lives in one 16-lane half)
#pragma unroll
    for (int r = 0; r < 8; r++) {
      float mx = s[0][r];
#pragma unroll
      for (int nt = 1; nt < 8; nt++) mx = fmaxf(mx, s[nt][r]);
      mx = hmax16(mx);
      float sum = 0.f;
#pragma unroll
      for (int nt = 0; nt < 8; nt++) {
        float e_ = __expf(s[nt][r] - mx);
        s[nt][r] = e_;
        sum += e_;
      }
      sum = hsum16(sum);
      float inv = 1.0f / sum;
#pragma unroll
      for (int nt = 0; nt < 8; nt++)
        swp[wid].P[r + 8 * half][nt * 16 + l16] = (bf16_t)(s[nt][r] * inv);
    }

    // o = P (16x128) @ V (128x32)  (8 WMMA)
    v8f o0 = vzero, o1 = vzero;
#pragma unroll
    for (int ks = 0; ks < 4; ks++) {
      v16bf ap;
      const bf16_t* src = &swp[wid].P[l16][ks * 32];
#pragma unroll
      for (int r = 0; r < 8; r++) {
        int kk = half * 16 + r * 2;
        ap[2 * r]     = src[kk];
        ap[2 * r + 1] = src[kk + 1];
      }
      v16bf bv0, bv1;
      const bf16_t* s0 = &sV[l16][ks * 32 + half * 16];
      const bf16_t* s1 = &sV[16 + l16][ks * 32 + half * 16];
#pragma unroll
      for (int e = 0; e < 16; e++) { bv0[e] = s0[e]; bv1[e] = s1[e]; }
      o0 = __builtin_amdgcn_wmma_f32_16x16x32_bf16(false, ap, false, bv0, (short)0, o0, false, false);
      o1 = __builtin_amdgcn_wmma_f32_16x16x32_bf16(false, ap, false, bv1, (short)0, o1, false, false);
    }

    // store in spatial order (folds _w2p)
#pragma unroll
    for (int r = 0; r < 8; r++) {
      int m   = qr0 + r + 8 * half;
      int t   = m >> 6, rem = m & 63;
      int ii  = rem >> 3, jj = rem & 7;
      long srow = ((long)(b * 2 + t) * HW + (wh * 8 + ii) * W + (ww * 8 + jj)) * HIDDEN + co;
      o_out[srow + l16]      = o0[r];
      o_out[srow + 16 + l16] = o1[r];
    }
  }
}

// ---------------------------------------------------------------------------
// LayerNorm + residual.  mode 0: out = x(remapped) + LN(f); mode 1: out += LN(f)
// ---------------------------------------------------------------------------
__global__ __launch_bounds__(256) void ln_res_kernel(
    const float* __restrict__ f, const float* __restrict__ gamma,
    const float* __restrict__ beta, const float* __restrict__ xres,
    float* __restrict__ out, int C, int mode, int rows_per_b, int stride_b,
    int base_off) {
  __shared__ float red1[256];
  __shared__ float red2[256];
  const int tid = threadIdx.x;
  const long r  = blockIdx.x;
  const float* frow = f + r * (long)C;
  float s = 0.f, s2 = 0.f;
  for (int j = tid; j < C; j += 256) {
    float v = frow[j];
    s += v; s2 += v * v;
  }
  red1[tid] = s; red2[tid] = s2;
  __syncthreads();
  for (int st = 128; st > 0; st >>= 1) {
    if (tid < st) { red1[tid] += red1[tid + st]; red2[tid] += red2[tid + st]; }
    __syncthreads();
  }
  const float mean = red1[0] / (float)C;
  const float var  = red2[0] / (float)C - mean * mean;
  const float inv  = rsqrtf(var + 1e-5f);
  const float* xr  = nullptr;
  if (mode == 0) {
    long bb = r / rows_per_b, rr = r % rows_per_b;
    xr = xres + (bb * stride_b + base_off + rr) * (long)C;
  }
  float* orow = out + r * (long)C;
  for (int j = tid; j < C; j += 256) {
    float v = (frow[j] - mean) * inv * gamma[j] + beta[j];
    orow[j] = (mode == 0 ? xr[j] : orow[j]) + v;
  }
}

// ---------------------------------------------------------------------------
// Host orchestration
// ---------------------------------------------------------------------------
extern "C" void kernel_launch(void* const* d_in, const int* in_sizes, int n_in,
                              void* d_out, int out_size, void* d_ws, size_t ws_size,
                              hipStream_t stream) {
  (void)in_sizes; (void)n_in; (void)out_size; (void)ws_size;
  const int Cs[3] = {256, 512, 1024};
  const int Hs[3] = {64, 32, 16};
  const float* feat[3] = {(const float*)d_in[0], (const float*)d_in[1], (const float*)d_in[2]};
  const float* position    = (const float*)d_in[3];
  const float* logit_scale = (const float*)d_in[4];
  const float* fcin[3][4]; const float* fcout[3][4]; const float* mlp[3][4];
  const float* nrm1[3][2]; const float* nrm2[3][2];
  for (int s = 0; s < 3; s++) {
    for (int j = 0; j < 4; j++) {
      fcin[s][j]  = (const float*)d_in[5 + s * 4 + j];
      fcout[s][j] = (const float*)d_in[17 + s * 4 + j];
      mlp[s][j]   = (const float*)d_in[29 + s * 4 + j];
    }
    for (int j = 0; j < 2; j++) {
      nrm1[s][j] = (const float*)d_in[41 + s * 2 + j];
      nrm2[s][j] = (const float*)d_in[47 + s * 2 + j];
    }
  }

  char* ws = (char*)d_ws;
  size_t off = 0;
  auto take = [&](size_t bytes) -> char* {
    char* p = ws + off;
    off += (bytes + 255) & ~(size_t)255;
    return p;
  };
  bf16_t *wfc1[3], *wfc2[3], *wfo1[3], *wfo2[3], *wm1[3], *wm2[3];
  for (int s = 0; s < 3; s++) {
    size_t c = Cs[s];
    wfc1[s] = (bf16_t*)take(c * 256 * 2);
    wfc2[s] = (bf16_t*)take((size_t)65536 * 2);
    wfo1[s] = (bf16_t*)take((size_t)65536 * 2);
    wfo2[s] = (bf16_t*)take(256 * c * 2);
    wm1[s]  = (bf16_t*)take(c * c * 2);
    wm2[s]  = (bf16_t*)take(c * c * 2);
  }
  float* buf0 = (float*)take((size_t)24576 * 256 * 4);   // h1p / h1f / fa
  float* buf1 = (float*)take((size_t)24576 * 256 * 4);   // pin / g1 / m2
  float* buf2 = (float*)take((size_t)16384 * 256 * 4);   // fin / o / m1
  bf16_t* qb  = (bf16_t*)take((size_t)16384 * 256 * 2);
  bf16_t* kb  = (bf16_t*)take((size_t)16384 * 256 * 2);
  bf16_t* vb  = (bf16_t*)take((size_t)16384 * 256 * 2);

  // convert+transpose all weights to bf16 W^T [N,K]
  auto cvtT = [&](const float* src, bf16_t* dst, int K, int N) {
    cvt_transpose_kernel<<<dim3(K / 32, N / 32), 256, 0, stream>>>(src, dst, K, N);
  };
  for (int s = 0; s < 3; s++) {
    int c = Cs[s];
    cvtT(fcin[s][0], wfc1[s], c, 256);
    cvtT(fcin[s][2], wfc2[s], 256, 256);
    cvtT(fcout[s][0], wfo1[s], 256, 256);
    cvtT(fcout[s][2], wfo2[s], 256, c);
    cvtT(mlp[s][0], wm1[s], c, c);
    cvtT(mlp[s][2], wm2[s], c, c);
  }

  float* outp = (float*)d_out;
  long out_off = 0;
  for (int s = 0; s < 3; s++) {
    const int c = Cs[s], H = Hs[s], W = Hs[s], HW = H * W;
    const int nw = (H / 8) * (W / 8);
    const int Mfp = 2 * 3 * HW;   // B * 3 frames
    const int Mff = 2 * 2 * HW;   // B * 2 frames
    const dim3 blk(256);

    // fcin on fp frames (0..2) -> pin
    gemm_kernel<true><<<dim3(Mfp / 128, 2), blk, 0, stream>>>(
        feat[s], wfc1[s], fcin[s][1], buf0, c, 256, 3 * HW, 4 * HW, 0);
    gemm_kernel<false><<<dim3(Mfp / 128, 2), blk, 0, stream>>>(
        buf0, wfc2[s], fcin[s][3], buf1, 256, 256, Mfp, Mfp, 0);
    // fcin on ff frames (2..3) -> fin
    gemm_kernel<true><<<dim3(Mff / 128, 2), blk, 0, stream>>>(
        feat[s], wfc1[s], fcin[s][1], buf0, c, 256, 2 * HW, 4 * HW, 2 * HW);
    gemm_kernel<false><<<dim3(Mff / 128, 2), blk, 0, stream>>>(
        buf0, wfc2[s], fcin[s][3], buf2, 256, 256, Mff, Mff, 0);

    // q/k/v window pack (bf16)
    {
      int tot = 2 * nw * 128 * (HIDDEN / 4);
      qkv_prep_kernel<<<(tot + 255) / 256, blk, 0, stream>>>(
          buf2, buf1, qb, kb, vb, nw, H, W);
    }
    // attention -> o (spatial order) into buf2
    attn_kernel<<<2 * nw * N_HEADS, 128, 0, stream>>>(
        qb, kb, vb, position, logit_scale, buf2, nw, H, W);

    // fcout
    gemm_kernel<true><<<dim3(Mff / 128, 2), blk, 0, stream>>>(
        buf2, wfo1[s], fcout[s][1], buf1, 256, 256, Mff, Mff, 0);
    gemm_kernel<false><<<dim3(Mff / 128, c / 128), blk, 0, stream>>>(
        buf1, wfo2[s], fcout[s][3], buf0, 256, c, Mff, Mff, 0);

    // out = x + LN(fa)
    float* outs = outp + out_off;
    ln_res_kernel<<<Mff, blk, 0, stream>>>(
        buf0, nrm1[s][0], nrm1[s][1], feat[s], outs, c, 0, 2 * HW, 4 * HW, 2 * HW);

    // out += LN(mlp2(out))
    gemm_kernel<true><<<dim3(Mff / 128, c / 128), blk, 0, stream>>>(
        outs, wm1[s], mlp[s][1], buf2, c, c, Mff, Mff, 0);
    gemm_kernel<false><<<dim3(Mff / 128, c / 128), blk, 0, stream>>>(
        buf2, wm2[s], mlp[s][3], buf1, c, c, Mff, Mff, 0);
    ln_res_kernel<<<Mff, blk, 0, stream>>>(
        buf1, nrm2[s][0], nrm2[s][1], nullptr, outs, c, 1, Mff, Mff, 0);

    out_off += (long)Mff * c;
  }
}
